// RnnDecoder_27676769255965
// MI455X (gfx1250) — matrix operand order repaired and checked by reading, not compiled, not run on hardware
//
#include <hip/hip_runtime.h>
#include <hip/hip_bf16.h>
#include <math.h>

// Problem dims (fixed by reference)
constexpr int Bv = 128;
constexpr int Lv = 512;
constexpr int Hv = 512;
constexpr int Tv = 1024;

constexpr int HPAD = 520;   // 512 + 8 halves pad -> row stride 1040B, avoids LDS bank clash
constexpr int WD   = 4;     // weight-stream software pipeline depth (tiles)

typedef __attribute__((ext_vector_type(16))) _Float16 v16h;
typedef __attribute__((ext_vector_type(8)))  _Float16 v8h;
typedef __attribute__((ext_vector_type(8)))  float    v8f;

// ---------------------------------------------------------------------------
// Kernel 1: pack W_hh (f32 [H][H], row-major, used as B = W_hh^T) into f16
// blocks laid out per-lane as this kernel's WMMA B operand (contiguous k, as in A):
//   block(ntile, kt): 32 lanes x 16 halves contiguous (1 KB per block).
//   lane L holds column n = ntile*16 + (L&15), K = kt*32 + (L>>4)*16 + j.
// ---------------------------------------------------------------------------
__global__ void pack_whh_kernel(const float* __restrict__ W_hh,
                                _Float16* __restrict__ Wp) {
    int tid = blockIdx.x * blockDim.x + threadIdx.x;   // 0 .. 512*512-1
    int j     =  tid        & 15;
    int lane  = (tid >> 4)  & 31;
    int kt    = (tid >> 9)  & 15;
    int ntile =  tid >> 13;
    int n = ntile * 16 + (lane & 15);
    int k = kt * 32 + (lane >> 4) * 16 + j;
    Wp[tid] = (_Float16)W_hh[n * Hv + k];
}

// ---------------------------------------------------------------------------
// Kernel 2: c = x @ W_ih^T + (b_ih + b_hh)   -> f32 [B][H] in workspace.
// ---------------------------------------------------------------------------
__global__ void input_proj_kernel(const float* __restrict__ x,
                                  const float* __restrict__ W_ih,
                                  const float* __restrict__ b_ih,
                                  const float* __restrict__ b_hh,
                                  float* __restrict__ c) {
    int idx = blockIdx.x * blockDim.x + threadIdx.x;   // 0 .. B*H-1
    int b = idx >> 9;
    int h = idx & (Hv - 1);
    const float* xr = x    + b * Lv;
    const float* wr = W_ih + h * Lv;
    float acc = b_ih[h] + b_hh[h];
    #pragma unroll 4
    for (int l = 0; l < Lv; l += 4) {
        float4 xv = *(const float4*)(xr + l);
        float4 wv = *(const float4*)(wr + l);
        acc += xv.x * wv.x + xv.y * wv.y + xv.z * wv.z + xv.w * wv.w;
    }
    c[idx] = acc;
}

// Hardware tanh (CDNA5 transcendental). v_nop satisfies the TRANS-op
// result-use hazard since the compiler can't schedule around inline asm.
__device__ __forceinline__ float hw_tanhf(float x) {
    float y;
    asm volatile("v_tanh_f32 %0, %1\n\tv_nop" : "=v"(y) : "v"(x));
    return y;
}

// Workgroup barrier that only drains DScnt (LDS h-stores), NOT STOREcnt:
// the global output stores are write-once and may float across steps.
__device__ __forceinline__ void lds_barrier() {
    asm volatile("s_wait_dscnt 0x0\n\t"
                 "s_barrier_signal -1\n\t"
                 "s_barrier_wait -1" ::: "memory");
}

// ---------------------------------------------------------------------------
// Kernel 3: the recurrence. grid = B/16 workgroups, 1024 threads (32 waves).
// Wave w owns output N-tile w (16 columns). h kept in LDS (f16, A-layout,
// double buffered). Per step: 16 chained v_wmma_f32_16x16x32_f16, with the
// B operands software-pipelined WD tiles deep from the L2-resident packed
// weight buffer (first WD tiles of step t+1 issued BEFORE the barrier so
// they land during stores/barrier), A tiles pipelined 1 deep from LDS,
// hardware v_tanh_f32, NT masked output store, one DS-only barrier per step.
// ---------------------------------------------------------------------------
__global__ __launch_bounds__(1024)
void rnn_scan_kernel(const float*    __restrict__ c,      // [B][H] f32
                     const _Float16* __restrict__ Wp,     // packed B operand
                     const int*      __restrict__ counts, // [B]
                     float*          __restrict__ out) {  // [B][T][H] f32
    __shared__ _Float16 hbuf[2][16][HPAD];

    const int tid   = threadIdx.x;
    const int wave  = tid >> 5;          // 0..31 == N-tile index
    const int lane  = tid & 31;
    const int g     = lane >> 4;         // half-wave group
    const int li    = lane & 15;
    const int bbase = blockIdx.x * 16;   // batch-row base
    const int col   = wave * 16 + li;    // output column owned by this lane

    // zero-init h0 buffer
    {
        _Float16* hz = &hbuf[0][0][0];
        for (int i = tid; i < 16 * HPAD; i += 1024) hz[i] = (_Float16)0.0f;
    }

    // c accumulator in C/D layout: VGPR r -> row m = r + 8*g, lane -> col
    v8f cacc;
    int cnt[8];
    #pragma unroll
    for (int r = 0; r < 8; ++r) {
        int m = r + 8 * g;
        cacc[r] = c[(bbase + m) * Hv + col];
        cnt[r]  = counts[bbase + m];
    }

    // Per-lane weight stream base: block(ntile=wave, kt) is 1 KB; this lane's
    // 32-byte chunk of kt sits at byte offset kt*1024 + lane*32.
    const _Float16* wlane = Wp + (size_t)wave * (16 * 32 * 16) + (size_t)lane * 16;

    __syncthreads();

    union Av { v16h v; v8h h[2]; };

    // Always-zero offset, re-defined by volatile asm once per step so the
    // weight loads are re-issued each step (no LICM promotion/spill) while
    // keeping kernarg provenance (=> GLOBAL, not FLAT, loads).
    int zoff = 0;

    // Prologue: first WD weight tiles of step 0 in flight.
    asm volatile("" : "+v"(zoff));
    v16h bpre[WD];
    #pragma unroll
    for (int i = 0; i < WD; ++i)
        bpre[i] = *(const v16h*)(wlane + zoff + (size_t)i * 512);

    for (int t = 0; t < Tv; ++t) {
        const int p = t & 1;
        v8f acc = cacc;
        const _Float16* wbase_t = wlane + zoff;       // this step's weights
        const _Float16* rowp    = &hbuf[p][li][0];    // this step's h (A)

        // Full-unroll SSA pipeline buffers.
        v16h bv[16];
        Av   av[16];
        #pragma unroll
        for (int i = 0; i < WD; ++i) bv[i] = bpre[i];
        av[0].h[0] = *(const v8h*)(rowp + g * 16);
        av[0].h[1] = *(const v8h*)(rowp + g * 16 + 8);

        #pragma unroll
        for (int kt = 0; kt < 16; ++kt) {
            if (kt + WD < 16)                          // B tile kt+WD in flight
                bv[kt + WD] = *(const v16h*)(wbase_t + (size_t)(kt + WD) * 512);
            if (kt + 1 < 16) {                         // A tile kt+1 in flight
                av[kt + 1].h[0] = *(const v8h*)(rowp + (kt + 1) * 32 + g * 16);
                av[kt + 1].h[1] = *(const v8h*)(rowp + (kt + 1) * 32 + g * 16 + 8);
            }
            acc = __builtin_amdgcn_wmma_f32_16x16x32_f16(
                      false, av[kt].v, false, bv[kt], (short)0, acc, false, false);
        }

        // tanh, masked NT output store, and f16 h into the other LDS buffer
        #pragma unroll
        for (int r = 0; r < 8; ++r) {
            const int m = r + 8 * g;
            float hv = hw_tanhf(acc[r]);
            float ov = (t < cnt[r]) ? hv : 0.0f;
            __builtin_nontemporal_store(ov,
                &out[((size_t)(bbase + m) * Tv + t) * Hv + col]);
            hbuf[p ^ 1][m][col] = (_Float16)hv;
        }

        // Issue next step's first WD weight tiles BEFORE the barrier: they
        // land while this wave drains DScnt / signals / waits.
        asm volatile("" : "+v"(zoff));
        const _Float16* wnext = wlane + zoff;
        #pragma unroll
        for (int i = 0; i < WD; ++i)
            bpre[i] = *(const v16h*)(wnext + (size_t)i * 512);

        lds_barrier();
    }
}

// ---------------------------------------------------------------------------
// Inputs (setup_inputs order): out_counts[B] i32, x[B*L] f32, W_ih[H*L] f32,
// W_hh[H*H] f32, b_ih[H] f32, b_hh[H] f32, max_len scalar i32.
// Workspace: [0, 512KB) packed f16 W_hh; [512KB, 768KB) f32 c.
// ---------------------------------------------------------------------------
extern "C" void kernel_launch(void* const* d_in, const int* in_sizes, int n_in,
                              void* d_out, int out_size, void* d_ws, size_t ws_size,
                              hipStream_t stream) {
    const int*   counts = (const int*)  d_in[0];
    const float* x      = (const float*)d_in[1];
    const float* W_ih   = (const float*)d_in[2];
    const float* W_hh   = (const float*)d_in[3];
    const float* b_ih   = (const float*)d_in[4];
    const float* b_hh   = (const float*)d_in[5];
    float*       out    = (float*)d_out;

    _Float16* Wp = (_Float16*)d_ws;
    float*    c  = (float*)((char*)d_ws + (size_t)Hv * Hv * sizeof(_Float16));

    pack_whh_kernel  <<<(Hv * Hv) / 256, 256, 0, stream>>>(W_hh, Wp);
    input_proj_kernel<<<(Bv * Hv) / 256, 256, 0, stream>>>(x, W_ih, b_ih, b_hh, c);
    rnn_scan_kernel  <<<Bv / 16, 1024, 0, stream>>>(c, Wp, counts, out);
}
